// SimpleQuantizer_15470472200272
// MI455X (gfx1250) — matrix operand order repaired and checked by reading, not compiled, not run on hardware
//
#include <hip/hip_runtime.h>
#include <stdint.h>

// ---------------- constants (from reference shapes) ----------------
#define BQ   16      // batch
#define DQ   256     // feature dim
#define TQ   2048    // time
#define QQ   8       // codebooks
#define KQ   1024    // codes per book
#define NROW (BQ * TQ)          // 32768 residual rows
#define WG_ROWS 128             // rows per workgroup (8 waves * 16)
#define ROW_PITCH 260           // 256 + 4 dword pad (bank-conflict avoidance)

typedef __attribute__((ext_vector_type(16))) __bf16 bf16x16;
typedef __attribute__((ext_vector_type(8)))  float  floatx8;

#if __has_builtin(__builtin_amdgcn_tensor_load_to_lds) && __has_builtin(__builtin_amdgcn_s_wait_tensorcnt)
#define USE_TDM 1
typedef __attribute__((ext_vector_type(4))) unsigned int tdm_u32x4;
typedef __attribute__((ext_vector_type(8))) int          tdm_i32x8;
typedef __attribute__((ext_vector_type(4))) int          tdm_i32x4;
#else
#define USE_TDM 0
#endif

__device__ __forceinline__ unsigned short f2bf(float f) {
    unsigned int u = __float_as_uint(f);
    unsigned int r = u + 0x7FFFu + ((u >> 16) & 1u);   // round-to-nearest-even
    return (unsigned short)(r >> 16);
}

#if USE_TDM
// One-shot TDM DMA: 8KB contiguous global -> LDS (1-D tile, 2048 x 4B elements).
// D# per CDNA5 ISA: group0 = {count|flags, lds_addr,
// global_addr[31:0], global_addr[56:32]|type=2}; group1 holds data_size,
// tensor_dim0/1, tile_dim0, tensor_dim0_stride. Groups 2/3 zero (<=2D).
// This toolchain uses the 6-arg builtin form:
//   (uint32x4 g0, int32x8 g1, int32x4, int32x4, int32x8, i32 cpol)
__device__ __forceinline__ void tdm_stage_8k(const void* gsrc, unsigned int lds_off) {
    unsigned long long ga = (unsigned long long)(uintptr_t)gsrc;
    tdm_u32x4 g0;
    g0[0] = 1u;                                   // count=1, user descriptor
    g0[1] = lds_off;                              // lds_addr (bytes)
    g0[2] = (unsigned int)ga;                     // global_addr[31:0]
    g0[3] = (unsigned int)(ga >> 32) | (2u << 30);// global_addr[56:32] | type=2
    tdm_i32x8 g1;
    g1[0] = 0x00020000;                           // workgroup_mask=0, data_size=2 (4B)
    g1[1] = (int)(2048u << 16);                   // tensor_dim0[15:0]=2048 in [31:16]
    g1[2] = (int)(1u << 16);                      // tensor_dim0 hi=0, tensor_dim1=1
    g1[3] = (int)(2048u << 16);                   // tile_dim0=2048 in [31:16]
    g1[4] = 0;                                    // tile_dim1=0 (1-D), tile_dim2=0
    g1[5] = 2048;                                 // tensor_dim0_stride = 2048
    g1[6] = 0;
    g1[7] = 0;
    tdm_i32x4 z4 = {0, 0, 0, 0};
    tdm_i32x8 z8 = {0, 0, 0, 0, 0, 0, 0, 0};
    __builtin_amdgcn_tensor_load_to_lds(g0, g1, z4, z4, z8, 0);
}
#endif

// ---------------- prepack: embed f32 -> B-fragment bf16 + 0.5*||e||^2 ----------------
// B-fragment layout per (q, tile, s): 32 lanes * 16 bf16 (32B per lane).
// ISA 16-bit B 32x16: lane<16 -> col=lane, K=0..7,16..23 ; lane>=16 -> col=lane-16, K=8..15,24..31.
__global__ void rvq_prepack_kernel(const float* __restrict__ embed,
                                   uint16_t* __restrict__ wsB,
                                   float* __restrict__ wsNorm) {
    int g = blockIdx.x * blockDim.x + threadIdx.x;   // q*K + c
    if (g >= QQ * KQ) return;
    int q = g >> 10;
    int c = g & (KQ - 1);
    int tile = c >> 4;
    int col  = c & 15;
    const float* e = embed + (size_t)g * DQ;
    size_t tb = (size_t)((q * 64 + tile) * 8) * 512;   // halfs: each s-block = 1024B = 512 halfs
    float nrm = 0.f;
    for (int d = 0; d < DQ; ++d) {
        float f = e[d];
        nrm += f * f;
        int s = d >> 5;
        int k = d & 31;
        int lanei = col + ((k & 8) ? 16 : 0);
        int j = (k & 7) + ((k & 16) ? 8 : 0);
        wsB[tb + (size_t)s * 512 + lanei * 16 + j] = f2bf(f);
    }
    wsNorm[g] = 0.5f * nrm;
}

// ---------------- main RVQ kernel ----------------
__global__ __launch_bounds__(256)
void rvq_main_kernel(const float* __restrict__ x,
                     const float* __restrict__ embed,
                     const unsigned char* __restrict__ wsB,
                     const float* __restrict__ wsNorm,
                     int* __restrict__ outIdx,
                     float* __restrict__ outY) {
    __shared__ float r_lds[WG_ROWS * ROW_PITCH];               // 130 KB residual (f32, exact)
    __shared__ __align__(32) unsigned char sB[2][8192];        // double-buffered B tile (16 codes x 256 d, bf16)

    const int tid  = threadIdx.x;
    const int lane = tid & 31;
    const int wave = tid >> 5;
    const int rowBase = blockIdx.x * WG_ROWS;
    const int bb = rowBase >> 11;        // batch index (128 | 2048 -> constant per WG)
    const int t0 = rowBase & (TQ - 1);   // time base

    // ---- load x slice -> residual in LDS (coalesced over t) ----
    {
        const int row = tid & 127;
        const int dof = tid >> 7;        // 0 or 1
        const float* xb = x + ((size_t)bb * DQ) * TQ + t0 + row;
        for (int d2 = 0; d2 < DQ; d2 += 2) {
            int d = d2 + dof;
            r_lds[row * ROW_PITCH + d] = xb[(size_t)d * TQ];
        }
    }
    __syncthreads();

    const int m16  = lane & 15;          // row within 16-row block / column within tile
    const int hsel = (lane >> 4) * 8;    // A-layout K-half select
#if USE_TDM
    const unsigned int ldsB0 = (unsigned int)(uintptr_t)&sB[0][0];
    const unsigned int ldsB1 = (unsigned int)(uintptr_t)&sB[1][0];
#endif

    for (int q = 0; q < QQ; ++q) {
        // ---- build A fragments (16 rows x 256 dims, bf16) for this wave ----
        bf16x16 afr[8];
        const float* rrow = &r_lds[(wave * 16 + m16) * ROW_PITCH];
        #pragma unroll
        for (int s = 0; s < 8; ++s) {
            const float* rp = rrow + s * 32 + hsel;
            union { unsigned short u[16]; bf16x16 v; } A;
            #pragma unroll
            for (int j = 0; j < 8; ++j) A.u[j] = f2bf(rp[j]);          // K = d0..d0+7
            #pragma unroll
            for (int j = 0; j < 8; ++j) A.u[8 + j] = f2bf(rp[16 + j]); // K = d0+16..d0+23
            afr[s] = A.v;
        }

        float best[8];
        int   bidx[8];
        #pragma unroll
        for (int v = 0; v < 8; ++v) { best[v] = 3.4e38f; bidx[v] = 0; }

        const unsigned char* gB = wsB + (size_t)q * 64 * 8192;

        // ---- stage tile 0 ----
#if USE_TDM
        if (wave == 0) {
            tdm_stage_8k(gB, ldsB0);
            __builtin_amdgcn_s_wait_tensorcnt(0);
        }
#else
        {
            const float4* src = (const float4*)(gB + (size_t)tid * 32);
            float4 p0 = src[0], p1 = src[1];
            float4* dst = (float4*)(&sB[0][tid * 32]);
            dst[0] = p0; dst[1] = p1;
        }
#endif
        __syncthreads();

        for (int t = 0; t < 64; ++t) {
            const int cur = t & 1;
            if (t + 1 < 64) {   // prefetch next tile into the other buffer
#if USE_TDM
                if (wave == 0)
                    tdm_stage_8k(gB + (size_t)(t + 1) * 8192, cur ? ldsB0 : ldsB1);
#else
                const float4* src = (const float4*)(gB + (size_t)(t + 1) * 8192 + (size_t)tid * 32);
                float4 p0 = src[0], p1 = src[1];
                float4* dst = (float4*)(&sB[cur ^ 1][tid * 32]);
                dst[0] = p0; dst[1] = p1;
#endif
            }
            floatx8 acc = {0.f, 0.f, 0.f, 0.f, 0.f, 0.f, 0.f, 0.f};
            #pragma unroll
            for (int s = 0; s < 8; ++s) {
                bf16x16 bfr = *(const bf16x16*)(&sB[cur][s * 1024 + lane * 32]);
                acc = __builtin_amdgcn_wmma_f32_16x16x32_bf16(
                    false, afr[s], false, bfr, (short)0, acc, false, false);
            }
            // score = 0.5*||e||^2 - <r,e>  (||r||^2 constant per row -> same argmin)
            float hn = wsNorm[q * KQ + t * 16 + m16];
            int kidx = t * 16 + m16;
            #pragma unroll
            for (int v = 0; v < 8; ++v) {
                float sv = hn - acc[v];
                bool lt = sv < best[v];                 // strict < keeps earliest (lowest) index
                best[v] = lt ? sv : best[v];
                bidx[v] = lt ? kidx : bidx[v];
            }
#if USE_TDM
            if (wave == 0 && t + 1 < 64)
                __builtin_amdgcn_s_wait_tensorcnt(0);   // DMA done before barrier releases readers
#endif
            __syncthreads();
        }

        // ---- cross-lane argmin within each 16-lane half (wave32 shuffles) ----
        #pragma unroll
        for (int v = 0; v < 8; ++v) {
            float bv = best[v]; int bi = bidx[v];
            #pragma unroll
            for (int off = 1; off < 16; off <<= 1) {
                float ov = __shfl_xor(bv, off, 32);
                int   oi = __shfl_xor(bi, off, 32);
                if (ov < bv || (ov == bv && oi < bi)) { bv = ov; bi = oi; }
            }
            best[v] = bv; bidx[v] = bi;
        }

        // ---- write indices: rows 0..7 from lane 0, rows 8..15 from lane 16 ----
        if (m16 == 0) {
            int half = (lane >> 4) * 8;
            #pragma unroll
            for (int v = 0; v < 8; ++v) {
                int row = wave * 16 + half + v;
                outIdx[(size_t)bb * QQ * TQ + q * TQ + t0 + row] = bidx[v];
            }
        }

        // ---- exact f32 residual update: r -= embed[q][code] ----
        #pragma unroll
        for (int m = 0; m < 16; ++m) {
            int srcl = (m < 8) ? 0 : 16;
            int code = __shfl(bidx[m & 7], srcl, 32);
            const float* ep = embed + ((size_t)q * KQ + code) * DQ + lane * 8;
            float* rp = &r_lds[(wave * 16 + m) * ROW_PITCH + lane * 8];
            float4 e0 = *(const float4*)ep;
            float4 e1 = *(const float4*)(ep + 4);
            float4 r0 = *(float4*)rp;
            float4 r1 = *(float4*)(rp + 4);
            r0.x -= e0.x; r0.y -= e0.y; r0.z -= e0.z; r0.w -= e0.w;
            r1.x -= e1.x; r1.y -= e1.y; r1.z -= e1.z; r1.w -= e1.w;
            *(float4*)rp = r0; *(float4*)(rp + 4) = r1;
        }
        __syncthreads();
    }

    // ---- y = x - r_final (exact reconstruction), coalesced over t ----
    {
        const int row = tid & 127;
        const int dof = tid >> 7;
        const float* xb = x    + ((size_t)bb * DQ) * TQ + t0 + row;
        float*       yb = outY + ((size_t)bb * DQ) * TQ + t0 + row;
        for (int d2 = 0; d2 < DQ; d2 += 2) {
            int d = d2 + dof;
            yb[(size_t)d * TQ] = xb[(size_t)d * TQ] - r_lds[row * ROW_PITCH + d];
        }
    }
}

// ---------------- host launch ----------------
extern "C" void kernel_launch(void* const* d_in, const int* in_sizes, int n_in,
                              void* d_out, int out_size, void* d_ws, size_t ws_size,
                              hipStream_t stream) {
    const float* x     = (const float*)d_in[0];   // [16, 256, 2048]
    const float* embed = (const float*)d_in[1];   // [8, 1024, 256]

    uint16_t* wsB    = (uint16_t*)d_ws;                                   // 4 MB bf16 B-fragments
    float*    wsNorm = (float*)((char*)d_ws + (size_t)QQ * KQ * DQ * 2);  // 32 KB half-norms

    int*   outIdx = (int*)d_out;                           // [16, 8, 2048] int32
    float* outY   = (float*)d_out + (size_t)BQ * QQ * TQ;  // [16, 256, 2048] f32

    rvq_prepack_kernel<<<(QQ * KQ + 255) / 256, 256, 0, stream>>>(embed, wsB, wsNorm);
    rvq_main_kernel<<<NROW / WG_ROWS, 256, 0, stream>>>(
        x, embed, (const unsigned char*)wsB, wsNorm, outIdx, outY);
}
